// STRITS_54589034332339
// MI455X (gfx1250) — matrix operand order repaired and checked by reading, not compiled, not run on hardware
//
#include <hip/hip_runtime.h>
#include <hip/hip_bf16.h>
#include <math.h>

// ---------------- problem constants ----------------
constexpr int cBS = 128, cT = 400, cP = 22, cF = 6, cR = 1024, cE = 128;
constexpr int cBP  = cBS * cP;     // 2816 rows
constexpr int cX   = cP * cF;      // 132
constexpr int cKIH = cF + cE;      // 134
constexpr int cKP  = 1184;         // padded K: 160 (cc+e+pad) + 1024 (h); multiple of 32
constexpr int cN4  = 4 * cR;       // 4096 gate columns
constexpr int cKT  = cKP / 32;     // 37 k-steps
constexpr int cBSTRIDE = 40;       // LDS B-tile row stride (ushorts), padded vs 32

// ---------------- WMMA types ----------------
typedef __attribute__((ext_vector_type(16))) __bf16        bf16x16;
typedef __attribute__((ext_vector_type(8)))  float         f32x8;
typedef __attribute__((ext_vector_type(4)))  unsigned int  u32x4;

union Frag { bf16x16 v; u32x4 q[2]; };

__device__ __forceinline__ unsigned short f2bf(float f) {
    unsigned int u = __float_as_uint(f);
    return (unsigned short)((u + 0x7FFFu + ((u >> 16) & 1u)) >> 16);
}

// ---------------- utility kernels ----------------
__global__ void zero_f32(float* __restrict__ p, int n) {
    int i = blockIdx.x * 256 + threadIdx.x;
    if (i < n) p[i] = 0.0f;
}

__global__ void zero_pad_act(unsigned short* __restrict__ actA) {
    // zero activation columns [134, 160) once per call (never written elsewhere)
    int i = blockIdx.x * 256 + threadIdx.x;
    const int npad = 160 - cKIH;                 // 26
    if (i < cBP * npad) {
        int bp = i / npad, k = i - bp * npad;
        actA[(size_t)bp * cKP + cKIH + k] = 0;
    }
}

__global__ void zero_accum(float* accum) {
    if (threadIdx.x == 0) { accum[0] = 0.0f; accum[1] = 0.0f; }
}

// Build bf16 [W_ih | 0 pad | W_hh] (4096 x 1184) and combined bias.
__global__ void prep_weights(const float* __restrict__ W_ih, const float* __restrict__ W_hh,
                             const float* __restrict__ b_ih, const float* __restrict__ b_hh,
                             unsigned short* __restrict__ Wb, float* __restrict__ bcomb) {
    int n = blockIdx.x;
    for (int k = threadIdx.x; k < cKP; k += 256) {
        float v;
        if (k < cKIH)      v = W_ih[(size_t)n * cKIH + k];
        else if (k < 160)  v = 0.0f;
        else               v = W_hh[(size_t)n * cR + (k - 160)];
        Wb[(size_t)n * cKP + k] = f2bf(v);
    }
    if (threadIdx.x == 0) bcomb[n] = b_ih[n] + b_hh[n];
}

// denominator = sum over (b,t,p,f<2) of (1 - mask)
__global__ void denom_kernel(const float* __restrict__ mask, float* __restrict__ accum) {
    __shared__ float red[256];
    int i = blockIdx.x * 256 + threadIdx.x;
    const int n = cBS * cT * cP * 2;
    float v = 0.0f;
    if (i < n) { int q = i >> 1, f = i & 1; v = 1.0f - mask[(size_t)q * cF + f]; }
    red[threadIdx.x] = v;
    __syncthreads();
    for (int off = 128; off > 0; off >>= 1) {
        if (threadIdx.x < off) red[threadIdx.x] += red[threadIdx.x + off];
        __syncthreads();
    }
    if (threadIdx.x == 0) atomicAdd(&accum[1], red[0]);
}

__global__ void final_loss(const float* __restrict__ accum, float* __restrict__ out) {
    if (threadIdx.x == 0) out[0] = accum[0] / accum[1];
}

// ---------------- per-step kernels ----------------
// 1) gamma_h decay: hdec = h * exp(-relu(d @ W_gh^T + b_gh)); also bf16 into act buffer
__global__ void __launch_bounds__(256) decay_kernel(
        const float* __restrict__ h, const float* __restrict__ delta,
        const float* __restrict__ W_gh, const float* __restrict__ b_gh,
        float* __restrict__ hdec, unsigned short* __restrict__ actA, int t) {
    int idx = blockIdx.x * 256 + threadIdx.x;
    if (idx >= cBP * cR) return;
    int bp = idx >> 10, r = idx & 1023;
    int b = bp / cP, p = bp - b * cP;
    const float* d = delta + (size_t)(b * cT + t) * cX + p * cF;
    const float* w = W_gh + (size_t)r * cF;
    float s = b_gh[r];
#pragma unroll
    for (int f = 0; f < cF; ++f) s += d[f] * w[f];
    float g  = __expf(-fmaxf(s, 0.0f));
    float hd = h[idx] * g;
    hdec[idx] = hd;
    actA[(size_t)bp * cKP + 160 + r] = f2bf(hd);
}

// 2) x_h = hdec @ W_hr^T + b_hr, then all F=6 feature math, pred + loss + act writes
__global__ void __launch_bounds__(256) feat_kernel(
        const float* __restrict__ hdec,
        const float* __restrict__ input, const float* __restrict__ target,
        const float* __restrict__ mask,  const float* __restrict__ delta,
        const float* __restrict__ st_embs,
        const float* __restrict__ W_gx, const float* __restrict__ b_gx,
        const float* __restrict__ W_hr, const float* __restrict__ b_hr,
        const float* __restrict__ W_fr, const float* __restrict__ b_fr,
        const float* __restrict__ W_wc, const float* __restrict__ b_wc,
        unsigned short* __restrict__ actA, float* __restrict__ pred,
        float* __restrict__ accum, int t) {
    __shared__ float red[cF][256];
    const int bp = blockIdx.x, tid = threadIdx.x;
    const int b = bp / cP, p = bp - b * cP;

    // convert e_t slice to bf16 activation columns [6,134)
    const float* e = st_embs + ((size_t)(b * cT + t) * cP + p) * cE;
    if (tid < cE) actA[(size_t)bp * cKP + cF + tid] = f2bf(e[tid]);

    // x_h partial reduction over R
    float s[cF];
#pragma unroll
    for (int f = 0; f < cF; ++f) s[f] = 0.0f;
    const float* hrow = hdec + (size_t)bp * cR;
    for (int r = tid; r < cR; r += 256) {
        float hv = hrow[r];
#pragma unroll
        for (int f = 0; f < cF; ++f) s[f] += hv * W_hr[(size_t)f * cR + r];
    }
#pragma unroll
    for (int f = 0; f < cF; ++f) red[f][tid] = s[f];
    __syncthreads();
    for (int off = 128; off > 0; off >>= 1) {
        if (tid < off) {
#pragma unroll
            for (int f = 0; f < cF; ++f) red[f][tid] += red[f][tid + off];
        }
        __syncthreads();
    }

    if (tid == 0) {
        const size_t base = (size_t)(b * cT + t) * cX + p * cF;
        float xv[cF], m[cF], dd[cF], xh[cF], xc[cF], gx[cF], zh[cF], bt[cF], ch[cF], cc[cF];
#pragma unroll
        for (int f = 0; f < cF; ++f) {
            xv[f] = input[base + f];
            m[f]  = mask[base + f];
            dd[f] = delta[base + f];
            xh[f] = red[f][0] + b_hr[f];
            gx[f] = __expf(-fmaxf(dd[f] * W_gx[f * cF + f] + b_gx[f], 0.0f));
            xc[f] = m[f] * xv[f] + (1.0f - m[f]) * xh[f];
        }
#pragma unroll
        for (int i = 0; i < cF; ++i) {
            float z = b_fr[i];
#pragma unroll
            for (int j = 0; j < cF; ++j) if (j != i) z += W_fr[i * cF + j] * xc[j];
            zh[i] = z;
            float bb = b_wc[i];
#pragma unroll
            for (int j = 0; j < cF; ++j)
                bb += W_wc[i * 2 * cF + j] * gx[j] + W_wc[i * 2 * cF + cF + j] * m[j];
            bt[i] = bb;
        }
        float num = 0.0f;
#pragma unroll
        for (int f = 0; f < cF; ++f) {
            ch[f] = bt[f] * zh[f] + (1.0f - bt[f]) * xh[f];
            cc[f] = m[f] * xv[f] + (1.0f - m[f]) * ch[f];
            pred[base + f] = cc[f];
            actA[(size_t)bp * cKP + f] = f2bf(cc[f]);
            if (f < 2) {
                float tg = target[base + f];
                float w  = 1.0f - m[f];
                num += w * (fabsf(xh[f] - tg) + fabsf(zh[f] - tg) + fabsf(ch[f] - tg));
            }
        }
        atomicAdd(&accum[0], num);
    }
}

// 3) gates(2816 x 4096) = act(2816 x 1184) @ Wb^T + bias   [bf16 WMMA, f32 acc]
//    block = 8 waves, 128x128 macro-tile. B tile (32x128) staged in LDS once per
//    k-step (shared by all 8 waves), double-buffered, padded stride vs bank conflicts.
//    Wave w computes rows [Mbase+16w, +16) x all 128 cols (8 accumulators).
__global__ void __launch_bounds__(256, 2) gates_gemm(
        const unsigned short* __restrict__ actA, const unsigned short* __restrict__ Wb,
        const float* __restrict__ bcomb, float* __restrict__ gates) {
    __shared__ __align__(16) unsigned short Bs[2][128 * cBSTRIDE];

    const int tid   = threadIdx.x;
    const int lane  = tid & 31;
    const int wave  = tid >> 5;
    const int Mbase = blockIdx.y * 128 + wave * 16;
    const int Nbase = blockIdx.x * 128;
    const int lr = lane & 15;   // A: row-in-tile, B: col-in-tile
    const int hi = lane >> 4;   // lane-half selector

    // cooperative B-tile loader: thread -> (column, K-half)
    const int bcol  = tid >> 1;             // 0..127
    const int bhalf = (tid & 1) * 16;       // 0 or 16 (ushorts)
    const unsigned short* bsrc = Wb + (size_t)(Nbase + bcol) * cKP + bhalf;
    const int bdstOff = bcol * cBSTRIDE + bhalf;

    f32x8 acc[8];
    const f32x8 zf = {0.f, 0.f, 0.f, 0.f, 0.f, 0.f, 0.f, 0.f};
#pragma unroll
    for (int i = 0; i < 8; ++i) acc[i] = zf;

    // A fragment: lanes 0-15 row=lr K{0..15}; lanes 16-31 row=lr K{16..31}
    const unsigned short* aRow = actA + (size_t)(Mbase + lr) * cKP + hi * 16;

    // preload k-step 0 B tile into buffer 0
    {
        u32x4 t0 = *(const u32x4*)(bsrc);
        u32x4 t1 = *(const u32x4*)(bsrc + 8);
        *(u32x4*)(&Bs[0][bdstOff])     = t0;
        *(u32x4*)(&Bs[0][bdstOff + 8]) = t1;
    }
    __syncthreads();

    for (int kt = 0; kt < cKT; ++kt) {
        const int kb  = kt * 32;
        const int buf = kt & 1;

        // stage next k-step's B tile into the other buffer (overlaps with WMMAs)
        if (kt + 1 < cKT) {
            u32x4 t0 = *(const u32x4*)(bsrc + kb + 32);
            u32x4 t1 = *(const u32x4*)(bsrc + kb + 40);
            *(u32x4*)(&Bs[buf ^ 1][bdstOff])     = t0;
            *(u32x4*)(&Bs[buf ^ 1][bdstOff + 8]) = t1;
        }

        Frag a;
        a.q[0] = *(const u32x4*)(aRow + kb);
        a.q[1] = *(const u32x4*)(aRow + kb + 8);
        __builtin_prefetch(aRow + kb + 32, 0, 0);   // global_prefetch next A line

#pragma unroll
        for (int nt = 0; nt < 8; ++nt) {
            // B fragment from LDS: col = nt*16+lr, contiguous K half per lane group
            const unsigned short* bp = &Bs[buf][(nt * 16 + lr) * cBSTRIDE + hi * 16];
            Frag bq;
            bq.q[0] = *(const u32x4*)(bp);
            bq.q[1] = *(const u32x4*)(bp + 8);
            acc[nt] = __builtin_amdgcn_wmma_f32_16x16x32_bf16(
                false, a.v, false, bq.v, (short)0, acc[nt], false, false);
        }
        __syncthreads();
    }

    // epilogue: C/D layout -> m = Mbase + v + 8*hi, n = Nbase + nt*16 + lr
    const int mrow = Mbase + hi * 8;
#pragma unroll
    for (int nt = 0; nt < 8; ++nt) {
        const int n = Nbase + nt * 16 + lr;
        const float bias = bcomb[n];
#pragma unroll
        for (int v = 0; v < 8; ++v)
            gates[(size_t)(mrow + v) * cN4 + n] = acc[nt][v] + bias;
    }
}

// 4) LSTM pointwise: gate order i,f,g,o (torch LSTMCell)
__global__ void __launch_bounds__(256) lstm_kernel(
        const float* __restrict__ gates, float* __restrict__ c, float* __restrict__ h) {
    int idx = blockIdx.x * 256 + threadIdx.x;
    if (idx >= cBP * cR) return;
    int bp = idx >> 10, r = idx & 1023;
    const float* g = gates + (size_t)bp * cN4;
    float ig = g[r], fg = g[cR + r], gg = g[2 * cR + r], og = g[3 * cR + r];
    float si = 1.0f / (1.0f + __expf(-ig));
    float sf = 1.0f / (1.0f + __expf(-fg));
    float so = 1.0f / (1.0f + __expf(-og));
    float cn = sf * c[idx] + si * tanhf(gg);
    float hn = so * tanhf(cn);
    c[idx] = cn;
    h[idx] = hn;
}

// ---------------- launcher ----------------
extern "C" void kernel_launch(void* const* d_in, const int* in_sizes, int n_in,
                              void* d_out, int out_size, void* d_ws, size_t ws_size,
                              hipStream_t stream) {
    (void)in_sizes; (void)n_in; (void)out_size; (void)ws_size;
    const float* input  = (const float*)d_in[0];
    const float* target = (const float*)d_in[1];
    const float* mask   = (const float*)d_in[2];
    const float* delta  = (const float*)d_in[3];
    const float* st     = (const float*)d_in[4];
    const float* W_gh   = (const float*)d_in[5];
    const float* b_gh   = (const float*)d_in[6];
    const float* W_gx   = (const float*)d_in[7];
    const float* b_gx   = (const float*)d_in[8];
    const float* W_hr   = (const float*)d_in[9];
    const float* b_hr   = (const float*)d_in[10];
    const float* W_fr   = (const float*)d_in[11];
    const float* b_fr   = (const float*)d_in[12];
    const float* W_wc   = (const float*)d_in[13];
    const float* b_wc   = (const float*)d_in[14];
    const float* W_ih   = (const float*)d_in[15];
    const float* W_hh   = (const float*)d_in[16];
    const float* b_ih   = (const float*)d_in[17];
    const float* b_hh   = (const float*)d_in[18];

    char* ws = (char*)d_ws;
    size_t off = 0;
    auto take = [&](size_t bytes) -> char* {
        char* p = ws + off;
        off += (bytes + 255) & ~(size_t)255;
        return p;
    };
    float*          h     = (float*)take((size_t)cBP * cR * 4);
    float*          c     = (float*)take((size_t)cBP * cR * 4);
    float*          hd    = (float*)take((size_t)cBP * cR * 4);
    float*          gates = (float*)take((size_t)cBP * cN4 * 4);
    unsigned short* actA  = (unsigned short*)take((size_t)cBP * cKP * 2);
    unsigned short* Wb    = (unsigned short*)take((size_t)cN4 * cKP * 2);
    float*          bcomb = (float*)take((size_t)cN4 * 4);
    float*          accum = (float*)take(256);

    float* out  = (float*)d_out;
    float* pred = out + 1;   // [loss, pred(BS,T,P*F)]

    const int nHR   = cBP * cR;
    const int gridE = (nHR + 255) / 256;

    // one-time per call (deterministic): init state + weights + denominator
    zero_f32<<<gridE, 256, 0, stream>>>(h, nHR);
    zero_f32<<<gridE, 256, 0, stream>>>(c, nHR);
    zero_pad_act<<<(cBP * (160 - cKIH) + 255) / 256, 256, 0, stream>>>(actA);
    zero_accum<<<1, 32, 0, stream>>>(accum);
    prep_weights<<<cN4, 256, 0, stream>>>(W_ih, W_hh, b_ih, b_hh, Wb, bcomb);
    denom_kernel<<<(cBS * cT * cP * 2 + 255) / 256, 256, 0, stream>>>(mask, accum);

    for (int t = 0; t < cT; ++t) {
        decay_kernel<<<gridE, 256, 0, stream>>>(h, delta, W_gh, b_gh, hd, actA, t);
        feat_kernel<<<cBP, 256, 0, stream>>>(hd, input, target, mask, delta, st,
                                             W_gx, b_gx, W_hr, b_hr, W_fr, b_fr,
                                             W_wc, b_wc, actA, pred, accum, t);
        gates_gemm<<<dim3(cN4 / 128, cBP / 128), 256, 0, stream>>>(actA, Wb, bcomb, gates);
        lstm_kernel<<<gridE, 256, 0, stream>>>(gates, c, h);
    }
    final_loss<<<1, 32, 0, stream>>>(accum, out);
}
